// CAAN_25074019074091
// MI455X (gfx1250) — matrix operand order repaired and checked by reading, not compiled, not run on hardware
//
#include <hip/hip_runtime.h>

typedef __attribute__((ext_vector_type(16))) _Float16 v16h;
typedef __attribute__((ext_vector_type(8)))  _Float16 v8h;
typedef __attribute__((ext_vector_type(8)))  float    v8f;

#define NROW 8192
#define KIN  1024
#define DIM  512
#define HDIM 256

// ---------- fragment helpers (CDNA5 WMMA f16 16x16x32 layouts) ----------

// A fragment: 16x32 (MxK) f16 from row-major matrix with leading dim ld.
__device__ __forceinline__ v16h load_a16(const _Float16* p, int lane, int ld,
                                         int row0, int k0) {
  const int row  = row0 + (lane & 15);
  const int half = lane >> 4;
  const _Float16* b0 = p + row * ld + k0 + half * 8;
  v8h lo = *(const v8h*)(b0);
  v8h hi = *(const v8h*)(b0 + 16);
  v16h a;
#pragma unroll
  for (int t = 0; t < 8; ++t) { a[t] = lo[t]; a[t + 8] = hi[t]; }
  return a;
}

// B fragment: 32x16 (KxN) f16 from an N-major (K-contiguous) matrix p[n][k].
__device__ __forceinline__ v16h load_b16(const _Float16* p, int lane, int ld,
                                         int n0, int k0) {
  const int row  = n0 + (lane & 15);
  const int half = lane >> 4;
  const _Float16* b0 = p + row * ld + k0 + half * 8;
  v8h lo = *(const v8h*)(b0);
  v8h hi = *(const v8h*)(b0 + 16);
  v16h b;
#pragma unroll
  for (int t = 0; t < 8; ++t) { b[t] = lo[t]; b[t + 8] = hi[t]; }
  return b;
}

__device__ __forceinline__ v8f wmma_f16(v16h a, v16h b, v8f c) {
  return __builtin_amdgcn_wmma_f32_16x16x32_f16(false, a, false, b, (short)0, c,
                                                false, false);
}

// half-wave (16 lane) reductions, wave32
__device__ __forceinline__ float hmax16(float v) {
#pragma unroll
  for (int m = 1; m <= 8; m <<= 1) v = fmaxf(v, __shfl_xor(v, m, 32));
  return v;
}
__device__ __forceinline__ float hsum16(float v) {
#pragma unroll
  for (int m = 1; m <= 8; m <<= 1) v += __shfl_xor(v, m, 32);
  return v;
}

// ---------- gfx1250 async global->LDS (16B per lane, ASYNCcnt tracked) ----
__device__ __forceinline__ void async_b128(unsigned int lds_off,
                                           const _Float16* g) {
  asm volatile("global_load_async_to_lds_b128 %0, %1, off"
               :: "v"(lds_off), "v"((unsigned long long)(size_t)g)
               : "memory");
}
__device__ __forceinline__ void wait_async0() {
#if __has_builtin(__builtin_amdgcn_s_wait_asynccnt)
  __builtin_amdgcn_s_wait_asynccnt(0);
#else
  asm volatile("s_wait_asynccnt 0" ::: "memory");
#endif
}

// ---------- conversion kernels ----------

__global__ void cvt_f32_f16(const float* __restrict__ in,
                            _Float16* __restrict__ out, int n) {
  int i = blockIdx.x * blockDim.x + threadIdx.x;
  if (i < n) out[i] = (_Float16)in[i];
}

// in: [K][O] f32 -> out: [O][K] f16 (K-contiguous)
__global__ void transpose_cvt(const float* __restrict__ in,
                              _Float16* __restrict__ out, int K, int O) {
  int i = blockIdx.x * blockDim.x + threadIdx.x;
  if (i < K * O) {
    int k = i / O, o = i % O;
    out[o * K + k] = (_Float16)in[k * O + o];
  }
}

// ---------- QKV projection: each wave computes a 16x64 strip ----------
// grid: (NROW/16, 2, 3), block 128 (4 waves)
__global__ void __launch_bounds__(128, 1)
qkv_kernel(const _Float16* __restrict__ xh,
           const _Float16* __restrict__ wqT, const _Float16* __restrict__ wkT,
           const _Float16* __restrict__ wvT,
           const float* __restrict__ bq, const float* __restrict__ bk,
           const float* __restrict__ bv,
           _Float16* __restrict__ qh, _Float16* __restrict__ kh,
           _Float16* __restrict__ vT) {
  const int lane  = threadIdx.x & 31;
  const int wave  = threadIdx.x >> 5;
  const int i0    = blockIdx.x * 16;
  const int o0    = (blockIdx.y * 4 + wave) * 64;
  const int which = blockIdx.z;

  const _Float16* wT  = (which == 0) ? wqT : (which == 1) ? wkT : wvT;
  const float*    bia = (which == 0) ? bq  : (which == 1) ? bk  : bv;

  v8f acc[4];
#pragma unroll
  for (int t = 0; t < 4; ++t) acc[t] = (v8f){};

#pragma unroll 2
  for (int k0 = 0; k0 < KIN; k0 += 32) {
    v16h a = load_a16(xh, lane, KIN, i0, k0);   // A reused across 4 B tiles
#pragma unroll
    for (int t = 0; t < 4; ++t) {
      v16h b = load_b16(wT, lane, KIN, o0 + t * 16, k0);
      acc[t] = wmma_f16(a, b, acc[t]);
    }
  }

  const int half8 = (lane >> 4) << 3;
#pragma unroll
  for (int t = 0; t < 4; ++t) {
    const int o    = o0 + t * 16 + (lane & 15);
    const float bb = bia[o];
    if (which == 0) {
      const float sc = 0.04419417382415922f;  // 1/sqrt(512) pre-scaled into q
#pragma unroll
      for (int e = 0; e < 8; ++e)
        qh[(size_t)(i0 + e + half8) * DIM + o] = (_Float16)((acc[t][e] + bb) * sc);
    } else if (which == 1) {
#pragma unroll
      for (int e = 0; e < 8; ++e)
        kh[(size_t)(i0 + e + half8) * DIM + o] = (_Float16)(acc[t][e] + bb);
    } else {
#pragma unroll
      for (int e = 0; e < 8; ++e)
        vT[(size_t)o * NROW + i0 + e + half8] = (_Float16)(acc[t][e] + bb);
    }
  }
}

// ---------- fused flash attention + FFN + score ----------
// grid: NROW/64 blocks, 128 threads (4 waves); each wave owns 16 query rows.
// LDS (64KB): kbuf[32][512] f16 (32KB) | vbuf[512][32] f16 (32KB)
//   - P transpose slices reuse kbuf after QK consumes it each iteration
//   - ctx staging reuses the whole 64KB after the j-loop
__global__ void __launch_bounds__(128, 1)
attn_ffn_kernel(const _Float16* __restrict__ qh, const _Float16* __restrict__ kh,
                const _Float16* __restrict__ vT, const _Float16* __restrict__ w1T,
                const float* __restrict__ b1, const float* __restrict__ w2,
                const float* __restrict__ b2, float* __restrict__ out) {
  __shared__ __align__(32) _Float16 lds[32768];
  _Float16* kbuf = lds;            // 16384 halves
  _Float16* vbuf = lds + 16384;    // 16384 halves
  const int lane = threadIdx.x & 31;
  const int wave = threadIdx.x >> 5;
  const int tid  = threadIdx.x;
  const int i0   = (blockIdx.x * 4 + wave) * 16;

  // resident q A-fragments (16 rows x 512, pre-scaled by 1/sqrt(d))
  v16h qf[16];
#pragma unroll
  for (int kk = 0; kk < 16; ++kk) qf[kk] = load_a16(qh, lane, DIM, i0, kk * 32);

  v8f acc[32];
#pragma unroll
  for (int c = 0; c < 32; ++c) acc[c] = (v8f){};

  float mrow[8], lrow[8];
#pragma unroll
  for (int e = 0; e < 8; ++e) { mrow[e] = -1e30f; lrow[e] = 0.f; }

  const unsigned int klds = (unsigned int)(size_t)kbuf;
  const unsigned int vlds = (unsigned int)(size_t)vbuf;

  for (int j0 = 0; j0 < NROW; j0 += 32) {
    __syncthreads();  // all waves done reading kbuf/vbuf from previous iter

    // --- async-fill k tile: rows j0..j0+15 of kh = contiguous 16KB ---
    const _Float16* ksrc = kh + (size_t)j0 * DIM;
#pragma unroll
    for (int i = 0; i < 8; ++i) {
      int off = (i * 128 + tid) * 8;                      // halves, 16B chunks
      async_b128(klds + (unsigned)off * 2, ksrc + off);
    }
    // --- async-fill v tile: vbuf[d][0..31] = vT[d][j0..j0+31], 512 rows ---
#pragma unroll
    for (int i = 0; i < 16; ++i) {
      int c   = i * 128 + tid;                            // 0..2047
      int d   = c >> 2;
      int seg = (c & 3) * 8;                              // halves
      async_b128(vlds + (unsigned)(d * 32 + seg) * 2,
                 vT + (size_t)d * NROW + j0 + seg);
    }
    wait_async0();
    __syncthreads();  // tiles visible to all waves

    // --- S = q @ k^T : 16 rows x 32 cols (two 16x16 tiles), k from LDS ---
    v8f s0 = {}, s1 = {};
#pragma unroll
    for (int kk = 0; kk < 16; ++kk) {
      v16h bk0 = load_b16(kbuf, lane, DIM, 0,  kk * 32);
      v16h bk1 = load_b16(kbuf, lane, DIM, 16, kk * 32);
      s0 = wmma_f16(qf[kk], bk0, s0);
      s1 = wmma_f16(qf[kk], bk1, s1);
    }
    __syncthreads();  // everyone done reading kbuf -> P slices may overwrite it

    // --- online softmax (row r = element r within a half-wave) ---
    float alpha[8];
#pragma unroll
    for (int e = 0; e < 8; ++e) {
      float t  = hmax16(fmaxf(s0[e], s1[e]));
      float mn = fmaxf(mrow[e], t);
      alpha[e] = __expf(mrow[e] - mn);
      mrow[e]  = mn;
      s0[e] = __expf(s0[e] - mn);
      s1[e] = __expf(s1[e] - mn);
      lrow[e] = lrow[e] * alpha[e] + hsum16(s0[e] + s1[e]);
    }
#pragma unroll
    for (int c = 0; c < 32; ++c)
#pragma unroll
      for (int e = 0; e < 8; ++e) acc[c][e] *= alpha[e];

    // --- transpose P (C-layout) -> A-layout via private slice in kbuf ---
    _Float16* pst = kbuf + wave * 512;  // 16x32 f16
#pragma unroll
    for (int e = 0; e < 8; ++e) {
      int R = e + ((lane >> 4) << 3);
      pst[R * 32 + (lane & 15)]      = (_Float16)s0[e];
      pst[R * 32 + 16 + (lane & 15)] = (_Float16)s1[e];
    }
    v16h pf = load_a16(pst, lane, 32, 0, 0);  // wave-private, DS in-order

    // --- ctx += P @ V, v from LDS ---
#pragma unroll
    for (int c = 0; c < 32; ++c) {
      v16h bv = load_b16(vbuf, lane, 32, c * 16, 0);
      acc[c] = wmma_f16(pf, bv, acc[c]);
    }
  }
  __syncthreads();  // j-loop done everywhere; reuse all 64KB for ctx staging

  // finalize ctx = acc / l, stage to LDS as f16 [16][512] per wave
  float inv[8];
#pragma unroll
  for (int e = 0; e < 8; ++e) inv[e] = 1.0f / lrow[e];
  _Float16* cslice = lds + wave * 8192;
#pragma unroll
  for (int c = 0; c < 32; ++c)
#pragma unroll
    for (int e = 0; e < 8; ++e) {
      int R = e + ((lane >> 4) << 3);
      cslice[R * DIM + c * 16 + (lane & 15)] = (_Float16)(acc[c][e] * inv[e]);
    }

  v16h cf[16];
#pragma unroll
  for (int kk = 0; kk < 16; ++kk) cf[kk] = load_a16(cslice, lane, DIM, 0, kk * 32);

  // h = relu(ctx @ W1 + b1); score = h @ W2 + b2
  float sp[8] = {};
  for (int oc = 0; oc < 16; ++oc) {
    const int o    = oc * 16 + (lane & 15);
    const float bb = b1[o];
    v8f h;
#pragma unroll
    for (int e = 0; e < 8; ++e) h[e] = bb;
#pragma unroll
    for (int kk = 0; kk < 16; ++kk) {
      v16h bw = load_b16(w1T, lane, DIM, oc * 16, kk * 32);
      h = wmma_f16(cf[kk], bw, h);
    }
    const float w2v = w2[o];
#pragma unroll
    for (int e = 0; e < 8; ++e) sp[e] += fmaxf(h[e], 0.f) * w2v;
  }
#pragma unroll
  for (int e = 0; e < 8; ++e) sp[e] = hsum16(sp[e]);

  if ((lane & 15) == 0) {
    const float bb2 = b2[0];
    const int rbase = i0 + ((lane >> 4) << 3);
#pragma unroll
    for (int e = 0; e < 8; ++e) out[rbase + e] = sp[e] + bb2;
  }
}

// ---------- launch ----------

extern "C" void kernel_launch(void* const* d_in, const int* in_sizes, int n_in,
                              void* d_out, int out_size, void* d_ws, size_t ws_size,
                              hipStream_t stream) {
  (void)in_sizes; (void)n_in; (void)out_size; (void)ws_size;
  const float* x  = (const float*)d_in[0];
  const float* Wq = (const float*)d_in[1];
  const float* bq = (const float*)d_in[2];
  const float* Wk = (const float*)d_in[3];
  const float* bk = (const float*)d_in[4];
  const float* Wv = (const float*)d_in[5];
  const float* bv = (const float*)d_in[6];
  const float* W1 = (const float*)d_in[7];
  const float* b1 = (const float*)d_in[8];
  const float* W2 = (const float*)d_in[9];
  const float* b2 = (const float*)d_in[10];

  char* ws = (char*)d_ws;
  size_t off = 0;
  _Float16* xh  = (_Float16*)(ws + off); off += (size_t)NROW * KIN * 2;
  _Float16* wqT = (_Float16*)(ws + off); off += (size_t)DIM  * KIN * 2;
  _Float16* wkT = (_Float16*)(ws + off); off += (size_t)DIM  * KIN * 2;
  _Float16* wvT = (_Float16*)(ws + off); off += (size_t)DIM  * KIN * 2;
  _Float16* w1T = (_Float16*)(ws + off); off += (size_t)HDIM * DIM * 2;
  _Float16* qh  = (_Float16*)(ws + off); off += (size_t)NROW * DIM * 2;
  _Float16* kh  = (_Float16*)(ws + off); off += (size_t)NROW * DIM * 2;
  _Float16* vT  = (_Float16*)(ws + off); off += (size_t)DIM  * NROW * 2;

  cvt_f32_f16<<<(NROW * KIN) / 256, 256, 0, stream>>>(x, xh, NROW * KIN);
  transpose_cvt<<<(KIN * DIM + 255) / 256, 256, 0, stream>>>(Wq, wqT, KIN, DIM);
  transpose_cvt<<<(KIN * DIM + 255) / 256, 256, 0, stream>>>(Wk, wkT, KIN, DIM);
  transpose_cvt<<<(KIN * DIM + 255) / 256, 256, 0, stream>>>(Wv, wvT, KIN, DIM);
  transpose_cvt<<<(DIM * HDIM + 255) / 256, 256, 0, stream>>>(W1, w1T, DIM, HDIM);

  qkv_kernel<<<dim3(NROW / 16, 2, 3), 128, 0, stream>>>(
      xh, wqT, wkT, wvT, bq, bk, bv, qh, kh, vT);

  attn_ffn_kernel<<<dim3(NROW / 64), 128, 0, stream>>>(
      qh, kh, vT, w1T, b1, W2, b2, (float*)d_out);
}
